// MyAttention_57226144252763
// MI455X (gfx1250) — matrix operand order repaired and checked by reading, not compiled, not run on hardware
//
#include <hip/hip_runtime.h>
#include <math.h>

typedef __attribute__((ext_vector_type(16))) _Float16 v16h;
typedef __attribute__((ext_vector_type(8)))  _Float16 v8h;
typedef __attribute__((ext_vector_type(8)))  float    v8f;

union AB16 { v16h v; v8h h[2]; };

#define DIMC     256
#define NHEADS   8
#define HDIM     32
#define NQ       4096
#define NBATCH   4
#define GW       32          // h = w = 64/SR = 32
#define NS       1024        // GW*GW
#define ATT_SCALE 0.17677669529663687f   // 32^-0.5

// ---------------- utility kernels ----------------

__global__ void k_zero(float* p, long n) {
    long i = (long)blockIdx.x * blockDim.x + threadIdx.x;
    if (i < n) p[i] = 0.0f;
}

__global__ void k_cvt(const float* __restrict__ s, _Float16* __restrict__ d, long n) {
    long i = (long)blockIdx.x * blockDim.x + threadIdx.x;
    if (i < n) d[i] = (_Float16)s[i];
}

// ---------------- token2map scatter ----------------
// one block (256 threads) per source token
__global__ void k_scatter(const float* __restrict__ xsrc, const float* __restrict__ csrc,
                          const float* __restrict__ loc,
                          float* __restrict__ feat, float* __restrict__ cnt,
                          float* __restrict__ csum) {
    int t = blockIdx.x;             // 0 .. B*N-1
    int c = threadIdx.x;            // channel
    float lx = loc[2 * t + 0], ly = loc[2 * t + 1];
    lx = fminf(fmaxf(lx, 0.0f), 1.0f) * (float)(GW - 1);
    ly = fminf(fmaxf(ly, 0.0f), 1.0f) * (float)(GW - 1);
    int ix = (int)rintf(lx);        // ties-to-even == jnp.round
    int iy = (int)rintf(ly);
    int b  = t >> 12;               // / 4096
    int cell = b * NS + iy * GW + ix;
    atomicAdd(&feat[(size_t)cell * DIMC + c], xsrc[(size_t)t * DIMC + c]);
    if (c == 0) {
        atomicAdd(&cnt[cell], 1.0f);
        atomicAdd(&csum[cell], csrc[t]);
    }
}

// one block per grid cell: divide by (count+eps), mask, emit f16 features + f32 conf
__global__ void k_finalize(const float* __restrict__ feat, const float* __restrict__ cnt,
                           const float* __restrict__ csum,
                           _Float16* __restrict__ xsg, float* __restrict__ conf) {
    int cell = blockIdx.x;
    int c = threadIdx.x;
    float cc  = cnt[cell];
    float inv = (cc > 0.0f) ? 1.0f / (cc + 1e-6f) : 0.0f;
    xsg[(size_t)cell * DIMC + c] = (_Float16)(feat[(size_t)cell * DIMC + c] * inv);
    if (c == 0) conf[cell] = csum[cell] * inv;
}

// ---------------- WMMA GEMM:  Y[M x 256] = A[M x 256] * W^T (+bias) ----------------
// One wave per 16(M) x 64(N) strip: A-fragment shared across 4 WMMAs per k-step.
// Outputs (any may be null): f32 row-major, f16 row-major, f16 transposed-per-batch
// (outHT[b][col][j], j = row % 1024) for attention-V consumption.
__global__ void k_gemm(const _Float16* __restrict__ A, const _Float16* __restrict__ W,
                       const float* __restrict__ bias,
                       float* __restrict__ outF, _Float16* __restrict__ outH,
                       _Float16* __restrict__ outHT) {
    int lane = threadIdx.x;
    int half = lane >> 4, lan = lane & 15;
    size_t m0 = (size_t)blockIdx.x * 16;
    size_t n0 = (size_t)blockIdx.y * 64;

    const _Float16* arow  = A + (m0 + lan) * DIMC + 16 * half;  // A-frag: 16 contiguous K halves
    const _Float16* wbase = W + (n0 + lan) * DIMC + 16 * half;  // B-frag: 16 contiguous K halves

    v8f acc0 = {}, acc1 = {}, acc2 = {}, acc3 = {};
    #pragma unroll
    for (int k0 = 0; k0 < DIMC; k0 += 32) {
        AB16 a, b0, b1, b2, b3;
        a.h[0]  = *(const v8h*)(arow + k0);
        a.h[1]  = *(const v8h*)(arow + k0 + 8);
        b0.h[0] = *(const v8h*)(wbase + (size_t) 0 * DIMC + k0);
        b0.h[1] = *(const v8h*)(wbase + (size_t) 0 * DIMC + k0 + 8);
        b1.h[0] = *(const v8h*)(wbase + (size_t)16 * DIMC + k0);
        b1.h[1] = *(const v8h*)(wbase + (size_t)16 * DIMC + k0 + 8);
        b2.h[0] = *(const v8h*)(wbase + (size_t)32 * DIMC + k0);
        b2.h[1] = *(const v8h*)(wbase + (size_t)32 * DIMC + k0 + 8);
        b3.h[0] = *(const v8h*)(wbase + (size_t)48 * DIMC + k0);
        b3.h[1] = *(const v8h*)(wbase + (size_t)48 * DIMC + k0 + 8);
        acc0 = __builtin_amdgcn_wmma_f32_16x16x32_f16(false, a.v, false, b0.v, (short)0, acc0, false, false);
        acc1 = __builtin_amdgcn_wmma_f32_16x16x32_f16(false, a.v, false, b1.v, (short)0, acc1, false, false);
        acc2 = __builtin_amdgcn_wmma_f32_16x16x32_f16(false, a.v, false, b2.v, (short)0, acc2, false, false);
        acc3 = __builtin_amdgcn_wmma_f32_16x16x32_f16(false, a.v, false, b3.v, (short)0, acc3, false, false);
    }

    float accs[4][8];
    #pragma unroll
    for (int r = 0; r < 8; r++) {
        accs[0][r] = acc0[r]; accs[1][r] = acc1[r];
        accs[2][r] = acc2[r]; accs[3][r] = acc3[r];
    }
    #pragma unroll
    for (int t = 0; t < 4; t++) {
        size_t n = n0 + t * 16 + lan;                 // C-layout column
        float bv = bias ? bias[n] : 0.0f;
        #pragma unroll
        for (int r = 0; r < 8; r++) {
            size_t row = m0 + r + 8 * half;           // C-layout row
            float v = accs[t][r] + bv;
            if (outF)  outF[row * DIMC + n] = v;
            if (outH)  outH[row * DIMC + n] = (_Float16)v;
            if (outHT) {                              // (batch, col, j) d-major layout
                size_t b = row >> 10, j = row & 1023;
                outHT[(b * DIMC + n) * NS + j] = (_Float16)v;
            }
        }
    }
}

// ---------------- layernorm (f32 in, f16 out), one block per row ----------------
__global__ void k_ln(const float* __restrict__ x, const float* __restrict__ g,
                     const float* __restrict__ b, _Float16* __restrict__ y) {
    __shared__ float red[DIMC];
    int row = blockIdx.x, c = threadIdx.x;
    float v = x[(size_t)row * DIMC + c];
    red[c] = v; __syncthreads();
    for (int s = 128; s > 0; s >>= 1) { if (c < s) red[c] += red[c + s]; __syncthreads(); }
    float mu = red[0] * (1.0f / DIMC); __syncthreads();
    float d = v - mu;
    red[c] = d * d; __syncthreads();
    for (int s = 128; s > 0; s >>= 1) { if (c < s) red[c] += red[c + s]; __syncthreads(); }
    float var = red[0] * (1.0f / DIMC);
    y[(size_t)row * DIMC + c] = (_Float16)(d * rsqrtf(var + 1e-5f) * g[c] + b[c]);
}

// ---------------- fused flash attention ----------------
// grid = (N/16, HEADS, B), block = 32 (one wave owns 16 queries x full 1024 keys).
// Vt is the d-major transposed V: Vt[b][c][j], so PV B-fragments load contiguously.
__global__ void k_attn(const _Float16* __restrict__ Q, const _Float16* __restrict__ Km,
                       const _Float16* __restrict__ Vt, const float* __restrict__ conf,
                       _Float16* __restrict__ O) {
    __shared__ _Float16 P[16 * 32];       // P-tile staging (C-layout -> A-layout)
    int lane = threadIdx.x;
    int half = lane >> 4, lan = lane & 15;
    int qt = blockIdx.x, h = blockIdx.y, b = blockIdx.z;

    // Q A-fragment (16 queries x 32 dims), loaded once
    const _Float16* qbase = Q + ((size_t)(b * NQ + qt * 16 + lan)) * DIMC + h * HDIM + 16 * half;
    AB16 qa;
    qa.h[0] = *(const v8h*)(qbase);
    qa.h[1] = *(const v8h*)(qbase + 8);

    // per-head transposed V base: rows are head-dims, cols are keys
    const _Float16* vt0 = Vt + ((size_t)b * DIMC + h * HDIM + lan) * NS;        // d = lan
    const _Float16* vt1 = Vt + ((size_t)b * DIMC + h * HDIM + 16 + lan) * NS;   // d = 16+lan

    float mrow[8], lrow[8], o0[8], o1[8];
    #pragma unroll
    for (int r = 0; r < 8; r++) { mrow[r] = -INFINITY; lrow[r] = 0.f; o0[r] = 0.f; o1[r] = 0.f; }

    const float* confb = conf + b * NS;

    for (int jt = 0; jt < NS; jt += 32) {
        // K^T B-fragments for two 16-key tiles (contiguous per lane)
        const _Float16* kbase = Km + ((size_t)(b * NS + jt + lan)) * DIMC + h * HDIM + 16 * half;
        AB16 kb0, kb1;
        kb0.h[0] = *(const v8h*)(kbase);
        kb0.h[1] = *(const v8h*)(kbase + 8);
        kb1.h[0] = *(const v8h*)(kbase + (size_t)16 * DIMC);
        kb1.h[1] = *(const v8h*)(kbase + (size_t)16 * DIMC + 8);

        v8f s0 = {}, s1 = {};
        s0 = __builtin_amdgcn_wmma_f32_16x16x32_f16(false, qa.v, false, kb0.v, (short)0, s0, false, false);
        s1 = __builtin_amdgcn_wmma_f32_16x16x32_f16(false, qa.v, false, kb1.v, (short)0, s1, false, false);

        float bias0 = confb[jt + lan];
        float bias1 = confb[jt + 16 + lan];

        float t0[8], t1[8], nm[8];
        #pragma unroll
        for (int r = 0; r < 8; r++) {
            t0[r] = s0[r] * ATT_SCALE + bias0;
            t1[r] = s1[r] * ATT_SCALE + bias1;
            float rm = fmaxf(t0[r], t1[r]);
            for (int m = 8; m >= 1; m >>= 1) rm = fmaxf(rm, __shfl_xor(rm, m, 16));
            nm[r] = fmaxf(mrow[r], rm);
        }
        #pragma unroll
        for (int r = 0; r < 8; r++) {
            float alpha = expf(mrow[r] - nm[r]);   // first iter: exp(-inf) = 0
            t0[r] = expf(t0[r] - nm[r]);
            t1[r] = expf(t1[r] - nm[r]);
            float rs = t0[r] + t1[r];
            for (int m = 8; m >= 1; m >>= 1) rs += __shfl_xor(rs, m, 16);
            lrow[r] = lrow[r] * alpha + rs;
            mrow[r] = nm[r];
            o0[r] *= alpha;
            o1[r] *= alpha;
        }
        __syncthreads();
        #pragma unroll
        for (int r = 0; r < 8; r++) {
            int row = r + 8 * half;                // rows owned by this lane in C-layout
            P[row * 32 + lan]      = (_Float16)t0[r];
            P[row * 32 + 16 + lan] = (_Float16)t1[r];
        }
        __syncthreads();

        // P A-fragment (16 queries x 32 keys)
        AB16 pa;
        pa.h[0] = *(const v8h*)(&P[lan * 32 + 16 * half]);
        pa.h[1] = *(const v8h*)(&P[lan * 32 + 16 * half + 8]);

        // V B-fragments from transposed V: lane holds 16 contiguous keys for its dim
        AB16 vb0, vb1;
        vb0.h[0] = *(const v8h*)(vt0 + jt + 16 * half);
        vb0.h[1] = *(const v8h*)(vt0 + jt + 16 * half + 8);
        vb1.h[0] = *(const v8h*)(vt1 + jt + 16 * half);
        vb1.h[1] = *(const v8h*)(vt1 + jt + 16 * half + 8);

        v8f a0 = {}, a1 = {};
        a0 = __builtin_amdgcn_wmma_f32_16x16x32_f16(false, pa.v, false, vb0.v, (short)0, a0, false, false);
        a1 = __builtin_amdgcn_wmma_f32_16x16x32_f16(false, pa.v, false, vb1.v, (short)0, a1, false, false);
        #pragma unroll
        for (int r = 0; r < 8; r++) { o0[r] += a0[r]; o1[r] += a1[r]; }
        __syncthreads();
    }

    #pragma unroll
    for (int r = 0; r < 8; r++) {
        float inv = 1.0f / lrow[r];
        size_t row = (size_t)(b * NQ + qt * 16 + r + 8 * half);
        O[row * DIMC + h * HDIM + lan]      = (_Float16)(o0[r] * inv);
        O[row * DIMC + h * HDIM + 16 + lan] = (_Float16)(o1[r] * inv);
    }
}

// ---------------- host orchestration ----------------

extern "C" void kernel_launch(void* const* d_in, const int* in_sizes, int n_in,
                              void* d_out, int out_size, void* d_ws, size_t ws_size,
                              hipStream_t stream) {
    const float* x    = (const float*)d_in[0];
    const float* xsrc = (const float*)d_in[1];
    const float* loc  = (const float*)d_in[2];
    const float* csrc = (const float*)d_in[3];
    const float* Wq   = (const float*)d_in[4];
    const float* Wk   = (const float*)d_in[5];
    const float* Wv   = (const float*)d_in[6];
    const float* Wsr  = (const float*)d_in[7];
    const float* b_sr = (const float*)d_in[8];
    const float* ln_g = (const float*)d_in[9];
    const float* ln_b = (const float*)d_in[10];
    const float* Wp   = (const float*)d_in[11];
    const float* bp   = (const float*)d_in[12];
    // d_in[13], d_in[14] = H, W (fixed 64 in this problem)

    const long MQ = (long)NBATCH * NQ;       // 16384 rows
    const long MS = (long)NBATCH * NS;       // 4096 rows
    const long NW = (long)DIMC * DIMC;       // 65536 weights

    char* w = (char*)d_ws;
    size_t off = 0;
    auto alloc = [&](size_t bytes) -> void* {
        void* p = w + off;
        off += (bytes + 255) & ~(size_t)255;
        return p;
    };
    _Float16* x16   = (_Float16*)alloc(MQ * DIMC * 2);
    _Float16* wq16  = (_Float16*)alloc(NW * 2);
    _Float16* wk16  = (_Float16*)alloc(NW * 2);
    _Float16* wv16  = (_Float16*)alloc(NW * 2);
    _Float16* wsr16 = (_Float16*)alloc(NW * 2);
    _Float16* wp16  = (_Float16*)alloc(NW * 2);
    float*    feat  = (float*)alloc(MS * DIMC * 4 + MS * 4 + MS * 4);  // feat|cnt|csum contiguous
    float*    cnt   = feat + MS * DIMC;
    float*    csum  = cnt + MS;
    _Float16* xsg16 = (_Float16*)alloc(MS * DIMC * 2);
    float*    conf  = (float*)alloc(MS * 4);
    float*    xs1   = (float*)alloc(MS * DIMC * 4);
    _Float16* xsl16 = (_Float16*)alloc(MS * DIMC * 2);
    _Float16* Q16   = (_Float16*)alloc(MQ * DIMC * 2);
    _Float16* K16   = (_Float16*)alloc(MS * DIMC * 2);
    _Float16* Vt16  = (_Float16*)alloc(MS * DIMC * 2);   // transposed per batch
    _Float16* Oat16 = (_Float16*)alloc(MQ * DIMC * 2);
    (void)ws_size; (void)in_sizes; (void)n_in; (void)out_size;

    // zero the scatter accumulators (contiguous region)
    long nz = MS * DIMC + 2 * MS;
    k_zero<<<dim3((nz + 255) / 256), 256, 0, stream>>>(feat, nz);

    // f32 -> f16 conversions
    k_cvt<<<dim3((MQ * DIMC + 255) / 256), 256, 0, stream>>>(x, x16, MQ * DIMC);
    k_cvt<<<dim3((NW + 255) / 256), 256, 0, stream>>>(Wq, wq16, NW);
    k_cvt<<<dim3((NW + 255) / 256), 256, 0, stream>>>(Wk, wk16, NW);
    k_cvt<<<dim3((NW + 255) / 256), 256, 0, stream>>>(Wv, wv16, NW);
    k_cvt<<<dim3((NW + 255) / 256), 256, 0, stream>>>(Wsr, wsr16, NW);
    k_cvt<<<dim3((NW + 255) / 256), 256, 0, stream>>>(Wp, wp16, NW);

    // token2map
    k_scatter<<<dim3(MQ), 256, 0, stream>>>(xsrc, csrc, loc, feat, cnt, csum);
    k_finalize<<<dim3(MS), 256, 0, stream>>>(feat, cnt, csum, xsg16, conf);

    // Q = x @ Wq^T  (f16 out for attention)
    k_gemm<<<dim3(MQ / 16, 4), 32, 0, stream>>>(x16, wq16, nullptr, nullptr, Q16, nullptr);
    // xs1 = token-map @ Wsr^T + b_sr (f32 for layernorm)
    k_gemm<<<dim3(MS / 16, 4), 32, 0, stream>>>(xsg16, wsr16, b_sr, xs1, nullptr, nullptr);
    k_ln<<<dim3(MS), DIMC, 0, stream>>>(xs1, ln_g, ln_b, xsl16);
    // K row-major, V transposed (d-major) for coalesced PV fragments
    k_gemm<<<dim3(MS / 16, 4), 32, 0, stream>>>(xsl16, wk16, nullptr, nullptr, K16, nullptr);
    k_gemm<<<dim3(MS / 16, 4), 32, 0, stream>>>(xsl16, wv16, nullptr, nullptr, nullptr, Vt16);

    // fused softmax(QK^T * scale + conf) @ V
    k_attn<<<dim3(NQ / 16, NHEADS, NBATCH), 32, 0, stream>>>(Q16, K16, Vt16, conf, Oat16);

    // final projection: out = attn_out @ Wp^T + bp  (f32 output)
    k_gemm<<<dim3(MQ / 16, 4), 32, 0, stream>>>(Oat16, wp16, bp, (float*)d_out, nullptr, nullptr);
}
